// TrafficGAT_27685359190741
// MI455X (gfx1250) — compile-verified, matrix operand order checked by reading, not run on hardware
//
#include <hip/hip_runtime.h>
#include <hip/hip_fp16.h>

// ---------------- problem constants (from reference) ----------------
#define Bb   8
#define Tt   12
#define Nn   1000
#define Ff   32
#define Ee   16000
#define Hh   64
#define Dd   24
#define Gg   (Bb * Tt)        // 96 graphs
#define NE   (Ee + Nn)        // 17000 edges incl. self loops
#define GN   (Gg * Nn)        // 96000 node rows
#define BN   (Bb * Nn)        // 8000 sequences
#define NEG_SLOPE 0.2f

typedef __attribute__((ext_vector_type(16))) _Float16 v16h;
typedef __attribute__((ext_vector_type(8)))  _Float16 v8h;
typedef __attribute__((ext_vector_type(8)))  float    v8f;

union F16x16 { v16h v; v8h p[2]; };

// =====================================================================
// WMMA GEMM:  C[M,Nv] (f32) = A[M,K] (f16 row major) * Bt[*,K]^T (f16)
// Bt stored row-major N x K (B transposed).  One 16 x (16*NT) strip per
// wave: NT accumulators, A fragment loaded once per K-step and reused.
// NT, K, ACC are compile-time so the K-loop fully unrolls with no
// control flow (no spills); only the ragged store mask is runtime.
// Requires: M % 64 == 0, Bt has >= gridDim.x*16*NT rows (zero-pad),
// Nv <= Bt rows (store/accumulate bound, may be ragged e.g. 24).
// =====================================================================
template <int NT, int K, bool ACC>
__global__ void k_gemm_wmma(const _Float16* __restrict__ A,
                            const _Float16* __restrict__ Bt,
                            float* __restrict__ C,
                            int Nv)
{
    const int lane = threadIdx.x & 31;
    const int wave = threadIdx.x >> 5;
    const int m0 = (blockIdx.y * 4 + wave) * 16;
    const int n0 = blockIdx.x * (16 * NT);

    const int lr  = lane & 15;
    const int khi = (lane >> 4) << 3;          // 0 or 8
    const _Float16* ap = A  + (size_t)(m0 + lr) * K + khi;
    const _Float16* bp = Bt + (size_t)(n0 + lr) * K + khi;   // tile j at +j*16*K

    const int mbase = m0 + ((lane >> 4) << 3);
    const v8f vzero = {0.f, 0.f, 0.f, 0.f, 0.f, 0.f, 0.f, 0.f};
    v8f acc[NT];
#pragma unroll
    for (int j = 0; j < NT; ++j) {
        if (ACC) {
            int ncol = n0 + j * 16 + lr;
#pragma unroll
            for (int r = 0; r < 8; ++r)
                acc[j][r] = (ncol < Nv) ? C[(size_t)(mbase + r) * Nv + ncol] : 0.f;
        } else {
            acc[j] = vzero;
        }
    }

#pragma unroll
    for (int kk = 0; kk < K; kk += 32) {
        F16x16 a;
        a.p[0] = *(const v8h*)(ap + kk);
        a.p[1] = *(const v8h*)(ap + kk + 16);
#pragma unroll
        for (int j = 0; j < NT; ++j) {
            F16x16 b;
            b.p[0] = *(const v8h*)(bp + (size_t)j * 16 * K + kk);
            b.p[1] = *(const v8h*)(bp + (size_t)j * 16 * K + kk + 16);
            acc[j] = __builtin_amdgcn_wmma_f32_16x16x32_f16(false, a.v, false, b.v,
                                                            (short)0, acc[j], false, false);
        }
    }

#pragma unroll
    for (int j = 0; j < NT; ++j) {
        int ncol = n0 + j * 16 + lr;
#pragma unroll
        for (int r = 0; r < 8; ++r)
            if (ncol < Nv) C[(size_t)(mbase + r) * Nv + ncol] = acc[j][r];
    }
}

// ---------------- elementwise / prep kernels ----------------
__global__ void k_cvt16(const float* __restrict__ s, _Float16* __restrict__ d, int n) {
    int i = blockIdx.x * blockDim.x + threadIdx.x;
    if (i < n) d[i] = (_Float16)s[i];
}
__global__ void k_fill16(_Float16* d, int n) {
    int i = blockIdx.x * blockDim.x + threadIdx.x;
    if (i < n) d[i] = (_Float16)0.f;
}
__global__ void k_fill32(float* d, float v, int n) {
    int i = blockIdx.x * blockDim.x + threadIdx.x;
    if (i < n) d[i] = v;
}
__global__ void k_fillu(unsigned* d, unsigned v, int n) {
    int i = blockIdx.x * blockDim.x + threadIdx.x;
    if (i < n) d[i] = v;
}
// Bt[n*K+k] = W[k*N+n]   (W is K x N, N==64 here)
__global__ void k_tr_cvt(const float* __restrict__ W, _Float16* __restrict__ Bt, int K) {
    int i = blockIdx.x * blockDim.x + threadIdx.x;
    if (i < K * 64) { int n = i / K, k = i % K; Bt[i] = (_Float16)W[k * 64 + n]; }
}
// conv weight (O=64, I=64, k=3) -> Bt[o][kk*64+ii]
__global__ void k_conv_wt(const float* __restrict__ Wc, _Float16* __restrict__ Bt) {
    int i = blockIdx.x * blockDim.x + threadIdx.x;
    if (i < 64 * 192) {
        int o = i / 192, r = i % 192, kk = r / 64, ii = r % 64;
        Bt[i] = (_Float16)Wc[(o * 64 + ii) * 3 + kk];
    }
}

// ---------------- GAT pieces ----------------
__global__ void k_gat_scores(const float* __restrict__ Hb,
                             const float* __restrict__ a_src,
                             const float* __restrict__ a_dst,
                             float* __restrict__ es, float* __restrict__ ed,
                             int heads, int cdim)
{
    int idx = blockIdx.x * blockDim.x + threadIdx.x;
    if (idx >= GN * heads) return;
    int h = idx % heads, gn = idx / heads;
    const float* hp = Hb + (size_t)gn * Hh + h * cdim;
    float s = 0.f, d = 0.f;
    for (int c = 0; c < cdim; ++c) { float v = hp[c]; s += v * a_src[h * cdim + c]; d += v * a_dst[h * cdim + c]; }
    es[idx] = s; ed[idx] = d;
}

__device__ __forceinline__ unsigned fflip(float f) {
    unsigned u = __float_as_uint(f);
    return (u & 0x80000000u) ? ~u : (u | 0x80000000u);
}
__device__ __forceinline__ float funflip(unsigned u) {
    return __uint_as_float((u & 0x80000000u) ? (u ^ 0x80000000u) : ~u);
}
__device__ __forceinline__ void edge_decode(int idx, int heads,
                                            const int* __restrict__ src,
                                            const int* __restrict__ dst,
                                            int& h, int& gs, int& gd)
{
    h = idx % heads;
    int t = idx / heads;
    int ei = t % NE;
    int g  = t / NE;
    int s, d;
    if (ei < Ee) { s = src[ei]; d = dst[ei]; } else { s = d = ei - Ee; }
    gs = g * Nn + s; gd = g * Nn + d;
}
__device__ __forceinline__ float edge_e(const float* es, const float* ed,
                                        int gs, int gd, int h, int heads) {
    float e = es[(size_t)gs * heads + h] + ed[(size_t)gd * heads + h];
    return e > 0.f ? e : NEG_SLOPE * e;
}

__global__ void k_edge_max(const float* __restrict__ es, const float* __restrict__ ed,
                           const int* __restrict__ src, const int* __restrict__ dst,
                           unsigned* __restrict__ mb, int heads)
{
    int idx = blockIdx.x * blockDim.x + threadIdx.x;
    if (idx >= Gg * NE * heads) return;
    int h, gs, gd; edge_decode(idx, heads, src, dst, h, gs, gd);
    atomicMax(&mb[(size_t)gd * heads + h], fflip(edge_e(es, ed, gs, gd, h, heads)));
}
__global__ void k_edge_sum(const float* __restrict__ es, const float* __restrict__ ed,
                           const int* __restrict__ src, const int* __restrict__ dst,
                           const unsigned* __restrict__ mb, float* __restrict__ sb, int heads)
{
    int idx = blockIdx.x * blockDim.x + threadIdx.x;
    if (idx >= Gg * NE * heads) return;
    int h, gs, gd; edge_decode(idx, heads, src, dst, h, gs, gd);
    float e = edge_e(es, ed, gs, gd, h, heads);
    float m = funflip(mb[(size_t)gd * heads + h]);
    atomicAdd(&sb[(size_t)gd * heads + h], __expf(e - m));
}
__global__ void k_init_out(float* __restrict__ Ob, const float* __restrict__ bias) {
    int i = blockIdx.x * blockDim.x + threadIdx.x;
    if (i < GN * Hh) Ob[i] = bias[i % Hh];
}
__global__ void k_edge_scatter(const float* __restrict__ es, const float* __restrict__ ed,
                               const int* __restrict__ src, const int* __restrict__ dst,
                               const unsigned* __restrict__ mb, const float* __restrict__ sb,
                               const float* __restrict__ Hb, float* __restrict__ Ob,
                               int heads, int cdim)
{
    int idx = blockIdx.x * blockDim.x + threadIdx.x;
    if (idx >= Gg * NE * heads) return;
    int h, gs, gd; edge_decode(idx, heads, src, dst, h, gs, gd);
    float e = edge_e(es, ed, gs, gd, h, heads);
    float m = funflip(mb[(size_t)gd * heads + h]);
    float alpha = __expf(e - m) / (sb[(size_t)gd * heads + h] + 1e-16f);
    const float* hp = Hb + (size_t)gs * Hh + h * cdim;
    float* op = Ob + (size_t)gd * Hh + h * cdim;
    for (int c = 0; c < cdim; ++c) atomicAdd(&op[c], hp[c] * alpha);
}
__global__ void k_elu_cvt(const float* __restrict__ s, _Float16* __restrict__ d, int n) {
    int i = blockIdx.x * blockDim.x + threadIdx.x;
    if (i < n) { float v = s[i]; v = v > 0.f ? v : (__expf(v) - 1.f); d[i] = (_Float16)v; }
}

// ---------------- temporal conv pieces ----------------
__global__ void k_im2col(const float* __restrict__ Ob, _Float16* __restrict__ A16) {
    int i = blockIdx.x * blockDim.x + threadIdx.x;
    if (i >= BN * Tt * 192) return;
    int f = i % 192; int row = i / 192;
    int t = row % Tt; int bn = row / Tt;
    int kk = f / 64, ii = f % 64;
    int ts = t + kk - 1;
    float v = 0.f;
    if (ts >= 0 && ts < Tt) {
        int b = bn / Nn, n = bn % Nn;
        v = Ob[((size_t)(b * Tt + ts) * Nn + n) * Hh + ii];
    }
    A16[i] = (_Float16)v;
}
__global__ void k_conv_bias_xs(const float* __restrict__ Cc, const float* __restrict__ bconv,
                               _Float16* __restrict__ xs) {
    int i = blockIdx.x * blockDim.x + threadIdx.x;
    if (i >= BN * Tt * Hh) return;
    int j = i % Hh; int row = i / Hh; int t = row % Tt; int bn = row / Tt;
    xs[((size_t)t * BN + bn) * Hh + j] = (_Float16)(Cc[i] + bconv[j]);
}

// ---------------- LSTM cell (gate order i,f,g,o) ----------------
__device__ __forceinline__ float sigm(float x) { return 1.f / (1.f + __expf(-x)); }
__global__ void k_lstm_cell(const float* __restrict__ gbuf,
                            const float* __restrict__ bih, const float* __restrict__ bhh,
                            float* __restrict__ cbuf, _Float16* __restrict__ h16,
                            _Float16* __restrict__ ys, int t)
{
    int i = blockIdx.x * blockDim.x + threadIdx.x;
    if (i >= BN * Hh) return;
    int j = i % Hh, bn = i / Hh;
    const float* g = gbuf + (size_t)bn * 256;
    float gi = g[j]        + bih[j]        + bhh[j];
    float gf = g[64 + j]   + bih[64 + j]   + bhh[64 + j];
    float gg = g[128 + j]  + bih[128 + j]  + bhh[128 + j];
    float go = g[192 + j]  + bih[192 + j]  + bhh[192 + j];
    float c = sigm(gf) * cbuf[i] + sigm(gi) * tanhf(gg);
    float h = sigm(go) * tanhf(c);
    cbuf[i] = c;
    _Float16 h16v = (_Float16)h;
    h16[i] = h16v;
    ys[(size_t)t * BN * Hh + i] = h16v;
}

// ---------------- MLP head ----------------
__global__ void k_relu_b_cvt(const float* __restrict__ C1, const float* __restrict__ b1,
                             _Float16* __restrict__ d) {
    int i = blockIdx.x * blockDim.x + threadIdx.x;
    if (i >= BN * 32) return;
    float v = C1[i] + b1[i % 32];
    d[i] = (_Float16)(v > 0.f ? v : 0.f);
}
__global__ void k_bias_out(const float* __restrict__ C2, const float* __restrict__ b2,
                           float* __restrict__ out) {
    int i = blockIdx.x * blockDim.x + threadIdx.x;
    if (i >= BN * Dd) return;
    out[i] = C2[i] + b2[i % Dd];
}

// =====================================================================
// host side
// =====================================================================
static inline unsigned gsz(long n, int b) { return (unsigned)((n + b - 1) / b); }

template <int K>
static void run_gat(hipStream_t st, const _Float16* A16in,
                    const float* W, const float* as, const float* ad, const float* bias,
                    _Float16* Wt, float* Hb, float* Ob, float* es, float* ed,
                    unsigned* mb, float* sb, const int* srcp, const int* dstp,
                    int heads, int cdim)
{
    k_tr_cvt<<<gsz((long)K * 64, 256), 256, 0, st>>>(W, Wt, K);
    dim3 gg(1, GN / 64);                                       // 16x64 strip per wave
    k_gemm_wmma<4, K, false><<<gg, 128, 0, st>>>(A16in, Wt, Hb, Hh);
    k_gat_scores<<<gsz((long)GN * heads, 256), 256, 0, st>>>(Hb, as, ad, es, ed, heads, cdim);
    k_fillu<<<gsz((long)GN * heads, 256), 256, 0, st>>>(mb, 0x007FFFFFu, GN * heads); // flip(-inf)
    k_fill32<<<gsz((long)GN * heads, 256), 256, 0, st>>>(sb, 0.f, GN * heads);
    long etot = (long)Gg * NE * heads;
    k_edge_max<<<gsz(etot, 256), 256, 0, st>>>(es, ed, srcp, dstp, mb, heads);
    k_edge_sum<<<gsz(etot, 256), 256, 0, st>>>(es, ed, srcp, dstp, mb, sb, heads);
    k_init_out<<<gsz((long)GN * Hh, 256), 256, 0, st>>>(Ob, bias);
    k_edge_scatter<<<gsz(etot, 256), 256, 0, st>>>(es, ed, srcp, dstp, mb, sb, Hb, Ob, heads, cdim);
}

extern "C" void kernel_launch(void* const* d_in, const int* in_sizes, int n_in,
                              void* d_out, int out_size, void* d_ws, size_t ws_size,
                              hipStream_t stream)
{
    (void)in_sizes; (void)n_in; (void)out_size; (void)ws_size;
    const float* x    = (const float*)d_in[0];
    const int*   srcp = (const int*)d_in[1];
    const int*   dstp = srcp + Ee;
    const float* g0W = (const float*)d_in[2],  *g0as = (const float*)d_in[3],
               * g0ad = (const float*)d_in[4], *g0b  = (const float*)d_in[5];
    const float* g1W = (const float*)d_in[6],  *g1as = (const float*)d_in[7],
               * g1ad = (const float*)d_in[8], *g1b  = (const float*)d_in[9];
    const float* g2W = (const float*)d_in[10], *g2as = (const float*)d_in[11],
               * g2ad = (const float*)d_in[12],*g2b  = (const float*)d_in[13];
    const float* convW = (const float*)d_in[14], *convb = (const float*)d_in[15];
    const float* Wih[2] = {(const float*)d_in[16], (const float*)d_in[20]};
    const float* Whh[2] = {(const float*)d_in[17], (const float*)d_in[21]};
    const float* bih[2] = {(const float*)d_in[18], (const float*)d_in[22]};
    const float* bhh[2] = {(const float*)d_in[19], (const float*)d_in[23]};
    const float* W1 = (const float*)d_in[24], *b1 = (const float*)d_in[25];
    const float* W2 = (const float*)d_in[26], *b2 = (const float*)d_in[27];
    float* out = (float*)d_out;

    // ---- workspace carve (256B aligned) ----
    char* ws = (char*)d_ws; size_t off = 0;
    auto alloc = [&](size_t bytes) -> char* {
        char* p = ws + off; off += (bytes + 255) & ~(size_t)255; return p;
    };
    _Float16* A16   = (_Float16*)alloc((size_t)GN * 192 * 2);   // GEMM A staging (max: im2col)
    float*    Hb    = (float*)   alloc((size_t)GN * Hh * 4);    // projected features / conv out
    float*    Ob    = (float*)   alloc((size_t)GN * Hh * 4);    // attention output
    float*    es    = (float*)   alloc((size_t)GN * 8 * 4);
    float*    ed    = (float*)   alloc((size_t)GN * 8 * 4);
    unsigned* mb    = (unsigned*)alloc((size_t)GN * 8 * 4);
    float*    sb    = (float*)   alloc((size_t)GN * 8 * 4);
    _Float16* WtG   = (_Float16*)alloc((size_t)64 * 64 * 2);    // GAT W^T (reused)
    _Float16* WtC   = (_Float16*)alloc((size_t)64 * 192 * 2);   // conv Bt
    _Float16* Wih16[2], *Whh16[2];
    for (int l = 0; l < 2; ++l) {
        Wih16[l] = (_Float16*)alloc((size_t)256 * 64 * 2);
        Whh16[l] = (_Float16*)alloc((size_t)256 * 64 * 2);
    }
    _Float16* W1_16 = (_Float16*)alloc((size_t)32 * 64 * 2);
    _Float16* W2_16 = (_Float16*)alloc((size_t)32 * 32 * 2);    // zero-padded 24->32 rows
    _Float16* xs16  = (_Float16*)alloc((size_t)GN * Hh * 2);    // [t][bn][64]
    _Float16* ysA   = (_Float16*)alloc((size_t)GN * Hh * 2);
    _Float16* ysB   = (_Float16*)alloc((size_t)GN * Hh * 2);
    float*    gbuf  = (float*)   alloc((size_t)BN * 256 * 4);
    float*    cbuf  = (float*)   alloc((size_t)BN * Hh * 4);
    _Float16* h16   = (_Float16*)alloc((size_t)BN * Hh * 2);
    float*    C1    = (float*)   alloc((size_t)BN * 32 * 4);
    float*    C2    = (float*)   alloc((size_t)BN * Dd * 4);

    // ---- GAT stack (96 graphs batched along M) ----
    k_cvt16<<<gsz((long)GN * Ff, 256), 256, 0, stream>>>(x, A16, GN * Ff);
    run_gat<Ff>(stream, A16, g0W, g0as, g0ad, g0b, WtG, Hb, Ob, es, ed, mb, sb, srcp, dstp, 8, 8);
    k_elu_cvt<<<gsz((long)GN * Hh, 256), 256, 0, stream>>>(Ob, A16, GN * Hh);
    run_gat<Hh>(stream, A16, g1W, g1as, g1ad, g1b, WtG, Hb, Ob, es, ed, mb, sb, srcp, dstp, 8, 8);
    k_elu_cvt<<<gsz((long)GN * Hh, 256), 256, 0, stream>>>(Ob, A16, GN * Hh);
    run_gat<Hh>(stream, A16, g2W, g2as, g2ad, g2b, WtG, Hb, Ob, es, ed, mb, sb, srcp, dstp, 1, 64);
    // Ob now holds hs [g=(b,t)][n][64], f32

    // ---- temporal conv (SAME, k=3) as im2col GEMM ----
    k_im2col<<<gsz((long)BN * Tt * 192, 256), 256, 0, stream>>>(Ob, A16);
    k_conv_wt<<<gsz(64L * 192, 256), 256, 0, stream>>>(convW, WtC);
    {
        dim3 gg(1, (BN * Tt) / 64);
        k_gemm_wmma<4, 192, false><<<gg, 128, 0, stream>>>(A16, WtC, Hb, Hh);
    }
    k_conv_bias_xs<<<gsz((long)BN * Tt * Hh, 256), 256, 0, stream>>>(Hb, convb, xs16);

    // ---- LSTM weights to f16 (Bt layout == Wih/Whh as stored) ----
    for (int l = 0; l < 2; ++l) {
        k_cvt16<<<gsz(256L * 64, 256), 256, 0, stream>>>(Wih[l], Wih16[l], 256 * 64);
        k_cvt16<<<gsz(256L * 64, 256), 256, 0, stream>>>(Whh[l], Whh16[l], 256 * 64);
    }

    // ---- 2-layer LSTM, T=12 ----
    const _Float16* xsrc[2] = {xs16, ysA};
    _Float16* ydst[2] = {ysA, ysB};
    for (int l = 0; l < 2; ++l) {
        k_fill16<<<gsz((long)BN * Hh, 256), 256, 0, stream>>>(h16, BN * Hh);
        k_fill32<<<gsz((long)BN * Hh, 256), 256, 0, stream>>>(cbuf, 0.f, BN * Hh);
        for (int t = 0; t < Tt; ++t) {
            dim3 gg(256 / 64, BN / 64);                        // 4 x 125
            k_gemm_wmma<4, 64, false><<<gg, 128, 0, stream>>>(
                xsrc[l] + (size_t)t * BN * Hh, Wih16[l], gbuf, 256);
            k_gemm_wmma<4, 64, true><<<gg, 128, 0, stream>>>(h16, Whh16[l], gbuf, 256);
            k_lstm_cell<<<gsz((long)BN * Hh, 256), 256, 0, stream>>>(
                gbuf, bih[l], bhh[l], cbuf, h16, ydst[l], t);
        }
    }

    // ---- MLP head on last hidden state ----
    k_cvt16<<<gsz(32L * 64, 256), 256, 0, stream>>>(W1, W1_16, 32 * 64);
    k_fill16<<<gsz(32L * 32, 256), 256, 0, stream>>>(W2_16, 32 * 32);
    k_cvt16<<<gsz(24L * 32, 256), 256, 0, stream>>>(W2, W2_16, 24 * 32);
    const _Float16* last = ysB + (size_t)(Tt - 1) * BN * Hh;
    {
        dim3 gg(1, BN / 64);
        k_gemm_wmma<2, 64, false><<<gg, 128, 0, stream>>>(last, W1_16, C1, 32);
    }
    k_relu_b_cvt<<<gsz((long)BN * 32, 256), 256, 0, stream>>>(C1, b1, A16);
    {
        dim3 gg(1, BN / 64);  // Nv=24 ragged store, B zero-padded to 32 rows
        k_gemm_wmma<2, 32, false><<<gg, 128, 0, stream>>>(A16, W2_16, C2, Dd);
    }
    k_bias_out<<<gsz((long)BN * Dd, 256), 256, 0, stream>>>(C2, b2, out);
}
